// _GPTJAttention_29635274342442
// MI455X (gfx1250) — matrix operand order repaired and checked by reading, not compiled, not run on hardware
//
#include <hip/hip_runtime.h>

#define S 2048
#define E 4096
#define H 16
#define D 256
#define RD 64

typedef __attribute__((ext_vector_type(16))) __bf16 v16bf;
typedef __attribute__((ext_vector_type(8)))  __bf16 v8bf;
typedef __attribute__((ext_vector_type(8)))  float  v8f;

__device__ __forceinline__ __bf16 f2bf(float f) {
  unsigned u = __float_as_uint(f);
  u += 0x7FFFu + ((u >> 16) & 1u);           // round-to-nearest-even
  unsigned short h = (unsigned short)(u >> 16);
  __bf16 r;
  __builtin_memcpy(&r, &h, sizeof(r));
  return r;
}

__device__ __forceinline__ v16bf cat16(v8bf lo, v8bf hi) {
  return __builtin_shufflevector(lo, hi, 0,1,2,3,4,5,6,7,8,9,10,11,12,13,14,15);
}

// ---------------------------------------------------------------- converts
__global__ void cvt_f32_to_bf16(const float* __restrict__ src,
                                __bf16* __restrict__ dst, int n) {
  int i = blockIdx.x * blockDim.x + threadIdx.x;
  if (i < n) dst[i] = f2bf(src[i]);
}

// fp32 [rows][cols] -> bf16 transposed [cols][rows], LDS-tiled (coalesced
// reads and writes). block (32,8), grid (cols/32, rows/32).
__global__ __launch_bounds__(256) void cvt_transpose_f32_bf16(
    const float* __restrict__ in, __bf16* __restrict__ out,
    int rows, int cols) {
  __shared__ float tile[32][33];
  const int c0 = blockIdx.x * 32;
  const int r0 = blockIdx.y * 32;
  const int tx = threadIdx.x;
  #pragma unroll
  for (int j = threadIdx.y; j < 32; j += 8)
    tile[j][tx] = in[(size_t)(r0 + j) * cols + c0 + tx];
  __syncthreads();
  #pragma unroll
  for (int j = threadIdx.y; j < 32; j += 8)
    out[(size_t)(c0 + j) * rows + r0 + tx] = f2bf(tile[tx][j]);
}

// bf16 [rows][cols] -> bf16 transposed [cols][rows], LDS-tiled.
__global__ __launch_bounds__(256) void transpose_bf16(
    const __bf16* __restrict__ in, __bf16* __restrict__ out,
    int rows, int cols) {
  __shared__ __bf16 tile[32][33];
  const int c0 = blockIdx.x * 32;
  const int r0 = blockIdx.y * 32;
  const int tx = threadIdx.x;
  #pragma unroll
  for (int j = threadIdx.y; j < 32; j += 8)
    tile[j][tx] = in[(size_t)(r0 + j) * cols + c0 + tx];
  __syncthreads();
  #pragma unroll
  for (int j = threadIdx.y; j < 32; j += 8)
    out[(size_t)(c0 + j) * rows + r0 + tx] = tile[tx][j];
}

// ------------------------------------------------- tiled WMMA bf16 GEMM
// C[M][N] = A[M][K] @ B, with B given pre-transposed as BT[N][K].
// Block: 128 threads = 4 waves; block tile 64(M)x64(N), wave 16(M)x64(N).
// A-fragments: contiguous 16B global loads (no intra-block A reuse).
// BT tile (shared by all 4 waves) staged in LDS with vector ds ops only.
template<bool OUT_BF16>
__global__ __launch_bounds__(128) void gemm_bf16_wmma(
    const __bf16* __restrict__ A, const __bf16* __restrict__ BT,
    void* __restrict__ Cout, int M, int N, int K) {
  __shared__ __bf16 Bs[64 * 32];   // [n][k]
  const int tid    = threadIdx.x;
  const int wave   = tid >> 5;
  const int lane   = tid & 31;
  const int lane16 = lane & 15;
  const int hi     = (lane >= 16);
  const int m0 = blockIdx.y * 64;
  const int n0 = blockIdx.x * 64;

  v8f acc[4];
  #pragma unroll
  for (int nt = 0; nt < 4; ++nt) acc[nt] = (v8f){};

  const int nrow_ld = tid >> 1;           // 0..63
  const int kcol_ld = (tid & 1) * 16;     // 0 / 16
  const __bf16* arow_p = A + (size_t)(m0 + wave * 16 + lane16) * K;
  const int aoff = hi ? 8 : 0;
  const int boff = hi ? 8 : 0;

  for (int k0 = 0; k0 < K; k0 += 32) {
    // stage BT 64x32 tile (coalesced 16B loads, vector LDS stores)
    const __bf16* bp = BT + (size_t)(n0 + nrow_ld) * K + k0 + kcol_ld;
    v8bf b_lo = *(const v8bf*)bp;
    v8bf b_hi = *(const v8bf*)(bp + 8);
    *(v8bf*)&Bs[nrow_ld * 32 + kcol_ld]     = b_lo;
    *(v8bf*)&Bs[nrow_ld * 32 + kcol_ld + 8] = b_hi;
    if (k0 + 32 < K) __builtin_prefetch(bp + 32, 0, 1);  // global_prefetch_b8
    __syncthreads();

    // A fragment: lanes 0-15 K 0-7 & 16-23; lanes 16-31 K 8-15 & 24-31
    v16bf afrag = cat16(*(const v8bf*)(arow_p + k0 + aoff),
                        *(const v8bf*)(arow_p + k0 + aoff + 16));
    #pragma unroll
    for (int nt = 0; nt < 4; ++nt) {
      const int bcol = nt * 16 + lane16;
      v16bf bfrag = cat16(*(const v8bf*)&Bs[bcol * 32 + boff],
                          *(const v8bf*)&Bs[bcol * 32 + boff + 16]);
      acc[nt] = __builtin_amdgcn_wmma_f32_16x16x32_bf16(
          false, afrag, false, bfrag, (short)0, acc[nt], false, false);
    }
    __syncthreads();
  }

  // C layout: VGPR r -> M=r (lanes 0-15) / r+8 (lanes 16-31), N=lane&15
  #pragma unroll
  for (int nt = 0; nt < 4; ++nt) {
    #pragma unroll
    for (int r = 0; r < 8; ++r) {
      const int row = m0 + wave * 16 + r + (hi ? 8 : 0);
      const int col = n0 + nt * 16 + lane16;
      float v = acc[nt][r];
      if (OUT_BF16) ((__bf16*)Cout)[(size_t)row * N + col] = f2bf(v);
      else          ((float*)Cout)[(size_t)row * N + col]  = v;
    }
  }
}

// ---------------------------------------------------------------- RoPE
__global__ void rope_convert(const float* __restrict__ qf,
                             const float* __restrict__ kf,
                             const int* __restrict__ pos,
                             __bf16* __restrict__ qb, __bf16* __restrict__ kb) {
  int idx = blockIdx.x * blockDim.x + threadIdx.x;
  if (idx >= S * E) return;
  int s = idx >> 12;          // / E
  int e = idx & (E - 1);
  int d = e & (D - 1);
  float q = qf[idx], k = kf[idx];
  if (d < RD) {
    float p    = (float)pos[s];
    int   i2   = d & ~1;
    float freq = powf(10000.0f, -(float)i2 / (float)RD);
    float ang  = p * freq;
    float sn = __sinf(ang), cs = __cosf(ang);
    float qp = qf[idx ^ 1];   // pair partner (idx parity == d parity)
    float kp = kf[idx ^ 1];
    float qr = (d & 1) ? qp : -qp;   // rotate_every_two
    float kr = (d & 1) ? kp : -kp;
    q = q * cs + qr * sn;
    k = k * cs + kr * sn;
  }
  qb[idx] = f2bf(q);
  kb[idx] = f2bf(k);
}

// ------------------------------------------- flash attention, 1 wave/block
// Wave handles (head h, 16-query block), streaming 32-key blocks with online
// softmax. Q frags live in registers; K B-frags are contiguous global loads
// (keys = N dim of row-major K); V is pre-transposed (vT[e][s]) so PV B-frags
// are contiguous global loads too. Only the 1KB P tile goes through LDS.
__global__ __launch_bounds__(32) void attn_flash_wmma(
    const __bf16* __restrict__ qb, const __bf16* __restrict__ kbuf,
    const __bf16* __restrict__ vT, __bf16* __restrict__ attb) {
  __shared__ __bf16 pl[16 * 32];   // P tile [row][key] (1 KB)
  const int lane   = threadIdx.x;
  const int lane16 = lane & 15;
  const int hi     = (lane >= 16);
  const int q0     = blockIdx.x * 16;
  const int h      = blockIdx.y;
  const int hbase  = h * D;
  const float scale = 0.0625f;     // 1/sqrt(256)

  // Q A-fragments for all 8 K-chunks of 32 (held in registers)
  v16bf qa[8];
  {
    const __bf16* qrow = qb + (size_t)(q0 + lane16) * E + hbase;
    #pragma unroll
    for (int c = 0; c < 8; ++c) {
      const int d0 = c * 32 + (hi ? 16 : 0);
      qa[c] = cat16(*(const v8bf*)(qrow + d0), *(const v8bf*)(qrow + d0 + 8));
    }
  }

  v8f o[16];
  #pragma unroll
  for (int nt = 0; nt < 16; ++nt) o[nt] = (v8f){};
  float mrow[8], lrow[8];
  #pragma unroll
  for (int r = 0; r < 8; ++r) { mrow[r] = -1e30f; lrow[r] = 0.0f; }

  const int qmax = q0 + 15;
  for (int kb0 = 0; kb0 <= qmax; kb0 += 32) {
    // scores: S = Q(16x256) @ K^T
    v8f s1 = (v8f){}, s2 = (v8f){};
    {
      const __bf16* k1 = kbuf + (size_t)(kb0 + lane16) * E + hbase;
      const __bf16* k2 = kbuf + (size_t)(kb0 + 16 + lane16) * E + hbase;
      #pragma unroll
      for (int c = 0; c < 8; ++c) {
        const int d0 = c * 32 + (hi ? 16 : 0);
        v16bf b1 = cat16(*(const v8bf*)(k1 + d0), *(const v8bf*)(k1 + d0 + 8));
        v16bf b2 = cat16(*(const v8bf*)(k2 + d0), *(const v8bf*)(k2 + d0 + 8));
        s1 = __builtin_amdgcn_wmma_f32_16x16x32_bf16(false, qa[c], false, b1,
                                                     (short)0, s1, false, false);
        s2 = __builtin_amdgcn_wmma_f32_16x16x32_bf16(false, qa[c], false, b2,
                                                     (short)0, s2, false, false);
      }
    }
    // online softmax (rows split across 16-lane groups; xor masks 1..8 stay
    // inside each group on wave32)
    #pragma unroll
    for (int r = 0; r < 8; ++r) {
      const int row  = q0 + r + (hi ? 8 : 0);
      const int key1 = kb0 + lane16;
      const int key2 = key1 + 16;
      float a = (key1 <= row) ? s1[r] * scale : -1e30f;   // causal mask
      float b = (key2 <= row) ? s2[r] * scale : -1e30f;
      float mx = fmaxf(a, b);
      #pragma unroll
      for (int msk = 1; msk < 16; msk <<= 1)
        mx = fmaxf(mx, __shfl_xor(mx, msk, 32));
      const float mn = fmaxf(mrow[r], mx);
      const float sc = __expf(mrow[r] - mn);
      const float p1 = __expf(a - mn);
      const float p2 = __expf(b - mn);
      float rs = p1 + p2;
      #pragma unroll
      for (int msk = 1; msk < 16; msk <<= 1)
        rs += __shfl_xor(rs, msk, 32);
      lrow[r] = lrow[r] * sc + rs;
      mrow[r] = mn;
      #pragma unroll
      for (int nt = 0; nt < 16; ++nt) o[nt][r] *= sc;
      const int prow = r + (hi ? 8 : 0);
      pl[prow * 32 + lane16]      = f2bf(p1);
      pl[prow * 32 + 16 + lane16] = f2bf(p2);
    }
    __syncthreads();
    // O += P(16x32) @ V(32x256): P re-read in A-layout from LDS, V B-frags
    // contiguous from vT[e][s].
    {
      const int poff = hi ? 16 : 0;
      v16bf pa = cat16(*(const v8bf*)&pl[lane16 * 32 + poff],
                       *(const v8bf*)&pl[lane16 * 32 + poff + 8]);
      const int koff = kb0 + (hi ? 16 : 0);
      #pragma unroll
      for (int nt = 0; nt < 16; ++nt) {
        const __bf16* vrow = vT + (size_t)(hbase + nt * 16 + lane16) * S + koff;
        v16bf bv = cat16(*(const v8bf*)vrow, *(const v8bf*)(vrow + 8));
        o[nt] = __builtin_amdgcn_wmma_f32_16x16x32_bf16(
            false, pa, false, bv, (short)0, o[nt], false, false);
      }
    }
    __syncthreads();
  }
  // normalize and emit bf16 attn output [S][E]
  #pragma unroll
  for (int nt = 0; nt < 16; ++nt) {
    #pragma unroll
    for (int r = 0; r < 8; ++r) {
      const int row = q0 + r + (hi ? 8 : 0);
      const int col = hbase + nt * 16 + lane16;
      attb[(size_t)row * E + col] = f2bf(o[nt][r] / lrow[r]);
    }
  }
}

// ---------------------------------------------------------------- launcher
extern "C" void kernel_launch(void* const* d_in, const int* in_sizes, int n_in,
                              void* d_out, int out_size, void* d_ws, size_t ws_size,
                              hipStream_t stream) {
  (void)in_sizes; (void)n_in; (void)out_size; (void)ws_size;
  const float* hs = (const float*)d_in[0];
  const float* wq = (const float*)d_in[1];
  const float* wk = (const float*)d_in[2];
  const float* wv = (const float*)d_in[3];
  const float* wo = (const float*)d_in[4];
  const int*  pos = (const int*)d_in[5];

  char* ws = (char*)d_ws;
  size_t off = 0;
  auto alloc = [&](size_t bytes) {
    char* p = ws + off;
    off += (bytes + 255) & ~(size_t)255;
    return (void*)p;
  };
  __bf16* hsb = (__bf16*)alloc((size_t)S * E * 2);
  __bf16* wqT = (__bf16*)alloc((size_t)E * E * 2);
  __bf16* wkT = (__bf16*)alloc((size_t)E * E * 2);
  __bf16* wvT = (__bf16*)alloc((size_t)E * E * 2);
  __bf16* woT = (__bf16*)alloc((size_t)E * E * 2);
  float*  qf  = (float*) alloc((size_t)S * E * 4);
  float*  kf  = (float*) alloc((size_t)S * E * 4);
  __bf16* qbb = (__bf16*)alloc((size_t)S * E * 2);
  __bf16* kbb = (__bf16*)alloc((size_t)S * E * 2);
  __bf16* vbb = (__bf16*)alloc((size_t)S * E * 2);
  __bf16* vT  = (__bf16*)alloc((size_t)S * E * 2);
  __bf16* atb = (__bf16*)alloc((size_t)S * E * 2);

  {
    int n = S * E;
    cvt_f32_to_bf16<<<(n + 255) / 256, 256, 0, stream>>>(hs, hsb, n);
  }
  // weights: fp32 [K][N] -> bf16 transposed [N][K]
  {
    dim3 tgrid(E / 32, E / 32), tblk(32, 8);
    cvt_transpose_f32_bf16<<<tgrid, tblk, 0, stream>>>(wq, wqT, E, E);
    cvt_transpose_f32_bf16<<<tgrid, tblk, 0, stream>>>(wk, wkT, E, E);
    cvt_transpose_f32_bf16<<<tgrid, tblk, 0, stream>>>(wv, wvT, E, E);
    cvt_transpose_f32_bf16<<<tgrid, tblk, 0, stream>>>(wo, woT, E, E);
  }

  dim3 ggrid(E / 64, S / 64);
  gemm_bf16_wmma<false><<<ggrid, 128, 0, stream>>>(hsb, wqT, qf,  S, E, E);
  gemm_bf16_wmma<false><<<ggrid, 128, 0, stream>>>(hsb, wkT, kf,  S, E, E);
  gemm_bf16_wmma<true ><<<ggrid, 128, 0, stream>>>(hsb, wvT, vbb, S, E, E);

  {
    int n = S * E;
    rope_convert<<<(n + 255) / 256, 256, 0, stream>>>(qf, kf, pos, qbb, kbb);
  }
  // V [S][E] -> vT [E][S]
  transpose_bf16<<<dim3(E / 32, S / 32), dim3(32, 8), 0, stream>>>(vbb, vT, S, E);

  attn_flash_wmma<<<dim3(S / 16, H), 32, 0, stream>>>(qbb, kbb, vT, atb);

  gemm_bf16_wmma<false><<<ggrid, 128, 0, stream>>>(atb, woT, (float*)d_out, S, E, E);
}
